// GCN_Encoder_70677981823585
// MI455X (gfx1250) — compile-verified, matrix operand order checked by reading, not run on hardware
//
#include <hip/hip_runtime.h>

#define N_NODES 50000
#define N_EDGES 800000
#define F_IN    128
#define HID     256
#define OUT_D   128
#define NPAD    50048   // N rounded up to multiple of 128

typedef __attribute__((ext_vector_type(16))) __bf16 v16bf;
typedef __attribute__((ext_vector_type(8)))  float  v8f;

union BFrag { unsigned u[8]; v16bf v; };

__device__ __forceinline__ unsigned short f2bf(float f) {
  unsigned u = __float_as_uint(f);
  u += 0x7FFFu + ((u >> 16) & 1u);   // round-to-nearest-even
  return (unsigned short)(u >> 16);
}

// Native CDNA5 f32 scatter-add: no-return atomic (STOREcnt), device scope (L2 RMW).
__device__ __forceinline__ void atomAddF(float* p, float v) {
  asm volatile("global_atomic_add_f32 %0, %1, off scope:SCOPE_DEV"
               :: "v"(p), "v"(v) : "memory");
}

// ---------------- degree / normalization ----------------
__global__ void k_deg_init(int* deg) {
  int i = blockIdx.x * 256 + threadIdx.x;
  if (i < N_NODES) deg[i] = 1;                       // self loop
}
__global__ void k_deg_edges(const int* __restrict__ row, int* deg) {
  int e = blockIdx.x * 256 + threadIdx.x;
  if (e < N_EDGES) atomicAdd(&deg[row[e]], 1);
}
__global__ void k_dis(float* dis_f, const int* deg) { // aliased buffers (per-thread RAW only)
  int i = blockIdx.x * 256 + threadIdx.x;
  if (i < N_NODES) dis_f[i] = rsqrtf((float)deg[i]);
}

// ---------------- conversions ----------------
__global__ void k_x_to_bf16(const float* __restrict__ x, unsigned short* __restrict__ xb) {
  long idx = (long)blockIdx.x * 256 + threadIdx.x;   // over NPAD*F_IN
  long i = idx >> 7;
  xb[idx] = (i < N_NODES) ? f2bf(x[idx]) : (unsigned short)0;
}

// dst[n*K+k] = bf16(src[(k0+k)*Nn + n])   (weight transposed to [N,K])
__global__ void k_wt(const float* __restrict__ src, unsigned short* __restrict__ dst,
                     int K, int Nn, int k0) {
  int idx = blockIdx.x * 256 + threadIdx.x;
  if (idx >= K * Nn) return;
  int n = idx / K, k = idx % K;
  dst[idx] = f2bf(src[(long)(k0 + k) * Nn + n]);
}

// zW[t][n] = sum_k z_emb[t][k] * W0[F_IN+k][n]   (grid=2, block=256)
__global__ void k_zW(const float* __restrict__ z_emb, const float* __restrict__ W0,
                     float* __restrict__ zW) {
  int t = blockIdx.x, n = threadIdx.x;
  float s = 0.f;
  for (int k = 0; k < HID; ++k) s += z_emb[t * HID + k] * W0[(long)(F_IN + k) * HID + n];
  zW[t * HID + n] = s;
}

__global__ void k_add_zW(float* hW, const int* __restrict__ z, const float* __restrict__ zW) {
  long idx = (long)blockIdx.x * 256 + threadIdx.x;   // over N*HID
  if (idx >= (long)N_NODES * HID) return;
  long i = idx >> 8; int f = idx & 255;
  hW[idx] += zW[z[i] * HID + f];
}

// ---------------- WMMA GEMM: C[NPAD,NN] = A[NPAD,K](bf16) * Bt[NN,K](bf16)^T ----------------
template<int K, int NN>
__global__ __launch_bounds__(256) void k_gemm(const unsigned short* __restrict__ A,
                                              const unsigned short* __restrict__ Bt,
                                              float* __restrict__ C) {
  constexpr int KS = (K > 128) ? 128 : K;     // LDS K-chunk (<=64KB staged B)
  constexpr int NT = NN / 16;
  extern __shared__ unsigned short blds[];    // [NN * KS] bf16, [n][k] layout
  const int tid  = threadIdx.x;
  const int wave = tid >> 5, lane = tid & 31;
  const int half = lane >> 4, l16 = lane & 15;
  const long row0 = (long)blockIdx.x * 128 + wave * 16;
  const long m    = row0 + l16;

  const v8f vzero = {};
  v8f acc[NT];
#pragma unroll
  for (int t = 0; t < NT; ++t) acc[t] = vzero;

  for (int ks = 0; ks < K; ks += KS) {
    __syncthreads();
#pragma unroll 1
    for (int v = tid; v < NN * KS / 8; v += 256) {  // stage B chunk, 16B coalesced
      ((uint4*)blds)[v] = *(const uint4*)(Bt + (long)(v / (KS / 8)) * K + ks + (v % (KS / 8)) * 8);
    }
    __syncthreads();
    for (int k0 = 0; k0 < KS; k0 += 32) {
      BFrag a;
#pragma unroll
      for (int j = 0; j < 8; ++j) {
        int ko = (j < 4) ? (half * 8 + 2 * j) : (16 + half * 8 + 2 * (j - 4));
        a.u[j] = *(const unsigned*)(A + m * K + ks + k0 + ko);
      }
#pragma unroll
      for (int t = 0; t < NT; ++t) {
        int n = t * 16 + l16;
        BFrag b;
#pragma unroll
        for (int j = 0; j < 8; ++j) {
          int ko = (j < 4) ? (half * 8 + 2 * j) : (16 + half * 8 + 2 * (j - 4));
          b.u[j] = *(const unsigned*)(blds + n * KS + k0 + ko);
        }
        acc[t] = __builtin_amdgcn_wmma_f32_16x16x32_bf16(
            false, a.v, false, b.v, (short)0, acc[t], false, false);
      }
    }
  }
#pragma unroll
  for (int t = 0; t < NT; ++t) {
    int n = t * 16 + l16;
#pragma unroll
    for (int r = 0; r < 8; ++r)
      C[(row0 + r + half * 8) * NN + n] = acc[t][r];
  }
}

// ---------------- aggregation ----------------
template<int D>
__global__ void k_agg_init(const float* __restrict__ dis, const float* __restrict__ hW,
                           const float* __restrict__ bias, float* __restrict__ out) {
  long idx = (long)blockIdx.x * 256 + threadIdx.x;   // over N*D
  if (idx >= (long)N_NODES * D) return;
  long i = idx / D; int f = idx % D;
  float d = dis[i];
  out[idx] = d * d * hW[idx] + bias[f];              // self-loop term + bias
}

// 4 features per thread: float4 gather (global_load_b128) + 4 native f32 atomics.
template<int D>
__global__ __launch_bounds__(256) void k_agg_edges(const int* __restrict__ row,
    const int* __restrict__ col, const float* __restrict__ dis,
    const float* __restrict__ hW, float* __restrict__ out) {
  constexpr int TPE = D / 4;            // threads per edge
  constexpr int EPB = 256 / TPE;        // edges per block
  const int s  = threadIdx.x / TPE;
  const int f4 = (threadIdx.x % TPE) * 4;
  long e = (long)blockIdx.x * EPB + s;
  if (e >= N_EDGES) return;
  int r = row[e], c = col[e];
  float wn = dis[r] * dis[c];
  const float4 hv = *(const float4*)(hW + (long)c * D + f4);
  float* o = out + (long)r * D + f4;
  atomAddF(o + 0, wn * hv.x);
  atomAddF(o + 1, wn * hv.y);
  atomAddF(o + 2, wn * hv.z);
  atomAddF(o + 3, wn * hv.w);
}

// ---------------- epilogues ----------------
__global__ __launch_bounds__(256) void k_ln_relu_bf16(const float* __restrict__ agg,
    const float* __restrict__ g, const float* __restrict__ bb,
    unsigned short* __restrict__ act) {
  const int i = blockIdx.x;       // over NPAD
  const int t = threadIdx.x;      // HID = 256
  if (i >= N_NODES) { act[(long)i * HID + t] = 0; return; }
  float v = agg[(long)i * HID + t];
  float s = v, q = v * v;
  for (int off = 16; off > 0; off >>= 1) {
    s += __shfl_down(s, off, 32);
    q += __shfl_down(q, off, 32);
  }
  __shared__ float s1[8], s2[8];
  __shared__ float mu_s, rstd_s;
  if ((t & 31) == 0) { s1[t >> 5] = s; s2[t >> 5] = q; }
  __syncthreads();
  if (t == 0) {
    float S = 0.f, Q = 0.f;
    for (int w = 0; w < 8; ++w) { S += s1[w]; Q += s2[w]; }
    float mu  = S / HID;
    float var = Q / HID - mu * mu;
    mu_s = mu; rstd_s = rsqrtf(var + 1e-5f);
  }
  __syncthreads();
  float y = (v - mu_s) * rstd_s * g[t] + bb[t];
  act[(long)i * HID + t] = f2bf(y > 0.f ? y : 0.f);
}

__global__ void k_relu_bf16(const float* __restrict__ agg, unsigned short* __restrict__ act) {
  long idx = (long)blockIdx.x * 256 + threadIdx.x;   // over NPAD*HID
  long i = idx >> 8;
  if (i < N_NODES) {
    float v = agg[idx];
    act[idx] = f2bf(v > 0.f ? v : 0.f);
  } else {
    act[idx] = 0;
  }
}

// ---------------- host ----------------
extern "C" void kernel_launch(void* const* d_in, const int* in_sizes, int n_in,
                              void* d_out, int out_size, void* d_ws, size_t ws_size,
                              hipStream_t stream) {
  const float* x     = (const float*)d_in[0];
  const int*   z     = (const int*)  d_in[1];
  const int*   erow  = (const int*)  d_in[2];
  const int*   ecol  = (const int*)  d_in[3];
  const float* z_emb = (const float*)d_in[4];
  const float* W0    = (const float*)d_in[5];
  const float* b0    = (const float*)d_in[6];
  const float* ln_g  = (const float*)d_in[7];
  const float* ln_b  = (const float*)d_in[8];
  const float* W1    = (const float*)d_in[9];
  const float* b1    = (const float*)d_in[10];
  const float* Wmu   = (const float*)d_in[11];
  const float* bmu   = (const float*)d_in[12];
  float* out = (float*)d_out;

  char* p = (char*)d_ws;
  float*          dis = (float*)p;           p += (size_t)NPAD * 4;
  unsigned short* xb  = (unsigned short*)p;  p += (size_t)NPAD * F_IN * 2;
  unsigned short* act = (unsigned short*)p;  p += (size_t)NPAD * HID * 2;
  float*          hW  = (float*)p;           p += (size_t)NPAD * HID * 4;
  float*          agg = (float*)p;           p += (size_t)NPAD * HID * 4;
  unsigned short* W0t = (unsigned short*)p;  p += 256 * 128 * 2;
  unsigned short* W1t = (unsigned short*)p;  p += 256 * 256 * 2;
  unsigned short* Wmt = (unsigned short*)p;  p += 256 * 128 * 2;
  float*          zW  = (float*)p;           p += 2 * 256 * 4;

  const int nb_nodes = (N_NODES + 255) / 256;
  const int nb_edges = (N_EDGES + 255) / 256;

  // normalization
  k_deg_init <<<nb_nodes, 256, 0, stream>>>((int*)dis);
  k_deg_edges<<<nb_edges, 256, 0, stream>>>(erow, (int*)dis);
  k_dis      <<<nb_nodes, 256, 0, stream>>>(dis, (const int*)dis);

  // conversions
  k_x_to_bf16<<<NPAD * F_IN / 256, 256, 0, stream>>>(x, xb);
  k_wt<<<(128 * 256 + 255) / 256, 256, 0, stream>>>(W0,  W0t, 128, 256, 0);
  k_wt<<<(256 * 256 + 255) / 256, 256, 0, stream>>>(W1,  W1t, 256, 256, 0);
  k_wt<<<(256 * 128 + 255) / 256, 256, 0, stream>>>(Wmu, Wmt, 256, 128, 0);
  k_zW<<<2, 256, 0, stream>>>(z_emb, W0, zW);

  // ---- layer 0: hW = x@W0[:128] + zW[z]; agg; LN; ReLU ----
  k_gemm<128, 256><<<NPAD / 128, 256, 256 * 128 * 2, stream>>>(xb, W0t, hW);
  k_add_zW<<<(N_NODES * HID + 255) / 256, 256, 0, stream>>>(hW, z, zW);
  k_agg_init<256><<<(N_NODES * HID + 255) / 256, 256, 0, stream>>>(dis, hW, b0, agg);
  k_agg_edges<256><<<(N_EDGES + 3) / 4, 256, 0, stream>>>(erow, ecol, dis, hW, agg);
  k_ln_relu_bf16<<<NPAD, 256, 0, stream>>>(agg, ln_g, ln_b, act);

  // ---- layer 1: hW = act@W1; agg; ReLU ----
  k_gemm<256, 256><<<NPAD / 128, 256, 256 * 128 * 2, stream>>>(act, W1t, hW);
  k_agg_init<256><<<(N_NODES * HID + 255) / 256, 256, 0, stream>>>(dis, hW, b1, agg);
  k_agg_edges<256><<<(N_EDGES + 3) / 4, 256, 0, stream>>>(erow, ecol, dis, hW, agg);
  k_relu_bf16<<<NPAD, 256, 0, stream>>>(agg, act);

  // ---- layer 2: hW = act@Wmu; agg -> d_out ----
  k_gemm<256, 128><<<NPAD / 128, 256, 128 * 128 * 2, stream>>>(act, Wmt, hW);
  k_agg_init<128><<<(N_NODES * OUT_D + 255) / 256, 256, 0, stream>>>(dis, hW, bmu, out);
  k_agg_edges<128><<<(N_EDGES + 7) / 8, 256, 0, stream>>>(erow, ecol, dis, hW, out);

  (void)in_sizes; (void)n_in; (void)out_size; (void)ws_size;
}
